// AttnBlock_1460288880886
// MI455X (gfx1250) — matrix operand order repaired and checked by reading, not compiled, not run on hardware
//
#include <hip/hip_runtime.h>
#include <hip/hip_bf16.h>

typedef __attribute__((ext_vector_type(16))) __bf16 v16bf;
typedef __attribute__((ext_vector_type(8)))  float  v8f;

static constexpr int Bn = 4;
static constexpr int C  = 512;
static constexpr int N  = 4096;   // 64 * 64
static constexpr int OC = 3 * C;  // 1536

#define DEVINL __device__ __forceinline__

DEVINL v8f wmma_bf16(v16bf a, v16bf b, v8f c) {
  // D = A(16x32 bf16) * B(32x16 bf16) + C(16x16 f32)
  return __builtin_amdgcn_wmma_f32_16x16x32_bf16(false, a, false, b,
                                                 (short)0, c, false, false);
}

union Frag {
  v16bf    v;
  unsigned u[8];
};

// A-matrix 16x32 fragment, source row-major bf16, lane holds row (pre-offset
// into row_base). k order used (contiguous, matches B): k = 2v + 16*half  (pairs).
DEVINL v16bf load_a_rowmajor_bf16(const __bf16* row_base, int k0, int half) {
  const unsigned* pu = (const unsigned*)(row_base + k0 + 16 * half);
  Frag f;
#pragma unroll
  for (int v = 0; v < 8; ++v) f.u[v] = pu[v];
  return f.v;
}

// B-matrix 32x16 fragment; source stored transposed (column-contiguous in k).
// k order used (matches A): k = 2v + 16*half  -> 16 contiguous bf16 per lane.
DEVINL v16bf load_b_rowmajor_bf16(const __bf16* col_base, int k0, int half) {
  const unsigned* pu = (const unsigned*)(col_base + k0 + 16 * half);
  Frag f;
#pragma unroll
  for (int v = 0; v < 8; ++v) f.u[v] = pu[v];
  return f.v;
}

// A-matrix fragment from fp32 LDS (probabilities P), converted to bf16.
DEVINL v16bf load_a_f32(const float* row_base, int k0, int half) {
  const float* p = row_base + k0 + 16 * half;
  v16bf r;
#pragma unroll
  for (int v = 0; v < 8; ++v) {
    int k = 2 * v;
    r[2 * v]     = (__bf16)p[k];
    r[2 * v + 1] = (__bf16)p[k + 1];
  }
  return r;
}

// ---------------------------------------------------------------------------
// Kernel 0: fp32 -> bf16 weight conversion (one-time, tiny).
// ---------------------------------------------------------------------------
__global__ void __launch_bounds__(256)
cvt_weights_kernel(const float* __restrict__ qkv_w, const float* __restrict__ proj_w,
                   __bf16* __restrict__ wqkv, __bf16* __restrict__ wproj) {
  int i = blockIdx.x * 256 + threadIdx.x;
  if (i < OC * C) wqkv[i] = (__bf16)qkv_w[i];
  if (i < C * C)  wproj[i] = (__bf16)proj_w[i];
}

// ---------------------------------------------------------------------------
// Kernel 1: qkv = qkv_w @ x + qkv_b   (per batch: [1536,512] x [512,4096])
// Emits Q,K as (b, n, c) bf16 and V as (b, c, n) bf16 (WMMA-friendly layouts).
// Each wave: 64(o) x 16(n) tile, K-loop over C in steps of 32.
// ---------------------------------------------------------------------------
__global__ void __launch_bounds__(256)
qkv_gemm_kernel(const float* __restrict__ x, const __bf16* __restrict__ wq,
                const float* __restrict__ bias, __bf16* __restrict__ Q,
                __bf16* __restrict__ K, __bf16* __restrict__ V) {
  const int lane = threadIdx.x & 31;
  const int wave = threadIdx.x >> 5;
  const int half = lane >> 4;
  const int mrow = lane & 15;
  const int b  = blockIdx.z;
  const int n0 = blockIdx.x * 16;
  const int o0 = blockIdx.y * 512 + wave * 64;

  const float* xb = x + (size_t)b * C * N;
  const int n = n0 + mrow;

  v8f acc[4];
#pragma unroll
  for (int t = 0; t < 4; ++t) acc[t] = (v8f){0, 0, 0, 0, 0, 0, 0, 0};

#pragma unroll 4
  for (int kc = 0; kc < C / 32; ++kc) {
    const int k0 = kc * 32;
    // B fragment from x (fp32 (c,n) layout -> strided over c, cvt to bf16)
    v16bf bf;
    {
      const float* p = xb + (size_t)(k0 + 16 * half) * N + n;
#pragma unroll
      for (int v = 0; v < 8; ++v) {
        bf[2 * v]     = (__bf16)p[(size_t)(2 * v) * N];
        bf[2 * v + 1] = (__bf16)p[(size_t)(2 * v + 1) * N];
      }
    }
#pragma unroll
    for (int t = 0; t < 4; ++t) {
      v16bf af =
          load_a_rowmajor_bf16(wq + (size_t)(o0 + t * 16 + mrow) * C, k0, half);
      acc[t] = wmma_bf16(af, bf, acc[t]);
    }
  }

#pragma unroll
  for (int t = 0; t < 4; ++t) {
#pragma unroll
    for (int r = 0; r < 8; ++r) {
      const int o = o0 + t * 16 + r + 8 * half;
      const __bf16 bv = (__bf16)(acc[t][r] + bias[o]);
      if (o < C)
        Q[(size_t)b * N * C + (size_t)n * C + o] = bv;
      else if (o < 2 * C)
        K[(size_t)b * N * C + (size_t)n * C + (o - C)] = bv;
      else
        V[(size_t)b * C * N + (size_t)(o - 2 * C) * N + n] = bv;
    }
  }
}

// ---------------------------------------------------------------------------
// Kernel 2: flash attention. One block = 16 query rows, 8 waves.
//  - Q row-block staged in LDS (bf16, padded)
//  - per 256-col j-chunk: waves compute S tile cooperatively (QK^T WMMA),
//    online softmax (running m,l + accumulator rescale), then P@V^T with
//    waves split over the C dimension (64 channels each, fp32 accumulators).
// Output Ht = hmat^T, layout (b, n, c) bf16.
// ---------------------------------------------------------------------------
__global__ void __launch_bounds__(256, 2)
attn_kernel(const __bf16* __restrict__ Q, const __bf16* __restrict__ K,
            const __bf16* __restrict__ V, __bf16* __restrict__ Ht) {
  constexpr int   CH   = 256;       // j-chunk width
  constexpr int   SLD  = CH + 4;    // padded fp32 row stride (bank-conflict free)
  constexpr float NEGI = -1.0e30f;
  __shared__ float    S[16 * SLD];      // 16.25 KB: score / P tile
  __shared__ unsigned Qs[16 * 260];     // 16.25 KB: Q rows, 256+4 dwords/row
  __shared__ float    red[16 * 17];
  __shared__ float    m_s[16], l_s[16], alpha_s[16];

  const int tid  = threadIdx.x;
  const int lane = tid & 31;
  const int wave = tid >> 5;
  const int half = lane >> 4;
  const int mrow = lane & 15;
  const int b  = blockIdx.y;
  const int i0 = blockIdx.x * 16;
  const float scale = 0.044194173824159216f;  // 512^-0.5

  const __bf16* Qb = Q + (size_t)b * N * C;
  const __bf16* Kb = K + (size_t)b * N * C;
  const __bf16* Vb = V + (size_t)b * C * N;

  {  // stage this block's 16 Q rows into padded LDS
    const unsigned* src = (const unsigned*)(Qb + (size_t)i0 * C);
    for (int idx = tid; idx < 16 * 256; idx += 256)
      Qs[(idx >> 8) * 260 + (idx & 255)] = src[idx];
  }
  if (tid < 16) { m_s[tid] = NEGI; l_s[tid] = 0.0f; }

  v8f acc[4];
#pragma unroll
  for (int t = 0; t < 4; ++t) acc[t] = (v8f){0, 0, 0, 0, 0, 0, 0, 0};

  const int row = tid & 15;
  const int c0  = tid >> 4;

  for (int chunk = 0; chunk < N / CH; ++chunk) {
    const int j0 = chunk * CH;
    __syncthreads();  // Q staged / previous PV finished reading S

    // ---- S = scale * Q K^T for this chunk: 2 j-tiles per wave ----
#pragma unroll
    for (int jt2 = 0; jt2 < 2; ++jt2) {
      const int jt   = wave * 2 + jt2;
      const int jcol = jt * 16 + mrow;
      const __bf16* kcol = Kb + (size_t)(j0 + jcol) * C;
      if (j0 + CH < N)  // warm L2/L0 for next chunk
        __builtin_prefetch(Kb + (size_t)(j0 + CH + jcol) * C, 0, 1);
      v8f s = (v8f){0, 0, 0, 0, 0, 0, 0, 0};
#pragma unroll 4
      for (int kc = 0; kc < C / 32; ++kc) {
        v16bf a  = load_a_rowmajor_bf16((const __bf16*)(Qs + mrow * 260),
                                        kc * 32, half);
        v16bf bb = load_b_rowmajor_bf16(kcol, kc * 32, half);
        s = wmma_bf16(a, bb, s);
      }
#pragma unroll
      for (int r = 0; r < 8; ++r)
        S[(r + 8 * half) * SLD + jt * 16 + mrow] = s[r] * scale;
    }
    __syncthreads();

    // ---- online softmax: chunk max -> running max / alpha ----
    {
      const float* sr = S + row * SLD + c0 * 16;
      float lmax = NEGI;
#pragma unroll
      for (int k2 = 0; k2 < 16; ++k2) lmax = fmaxf(lmax, sr[k2]);
      red[row * 17 + c0] = lmax;
    }
    __syncthreads();
    if (tid < 16) {
      float cm = NEGI;
#pragma unroll
      for (int k2 = 0; k2 < 16; ++k2) cm = fmaxf(cm, red[tid * 17 + k2]);
      const float mo = m_s[tid];
      const float mn = fmaxf(mo, cm);
      const float al = __expf(mo - mn);
      m_s[tid] = mn;
      alpha_s[tid] = al;
      l_s[tid] *= al;
    }
    __syncthreads();

    // rescale running output accumulators by alpha(row)
#pragma unroll
    for (int r = 0; r < 8; ++r) {
      const float al = alpha_s[r + 8 * half];
#pragma unroll
      for (int t = 0; t < 4; ++t) acc[t][r] *= al;
    }
    // exponentiate in place + partial row sums
    {
      const float mn = m_s[row];
      float* sr = S + row * SLD + c0 * 16;
      float ls = 0.0f;
#pragma unroll
      for (int k2 = 0; k2 < 16; ++k2) {
        const float p = __expf(sr[k2] - mn);
        sr[k2] = p;
        ls += p;
      }
      red[row * 17 + c0] = ls;
    }
    __syncthreads();
    if (tid < 16) {
      float ls = 0.0f;
#pragma unroll
      for (int k2 = 0; k2 < 16; ++k2) ls += red[tid * 17 + k2];
      l_s[tid] += ls;
    }

    // ---- acc += P @ V^T ; wave owns channels [wave*64, wave*64+64) ----
    const int cbase = wave * 64;
#pragma unroll 2
    for (int jt = 0; jt < CH / 32; ++jt) {
      v16bf a = load_a_f32(S + mrow * SLD, jt * 32, half);
#pragma unroll
      for (int t = 0; t < 4; ++t) {
        const __bf16* vrow = Vb + (size_t)(cbase + t * 16 + mrow) * N;
        if (jt == 0 && j0 + CH < N)
          __builtin_prefetch(vrow + j0 + CH, 0, 1);
        v16bf bb = load_b_rowmajor_bf16(vrow, j0 + jt * 32, half);
        acc[t] = wmma_bf16(a, bb, acc[t]);
      }
    }
  }
  __syncthreads();

  // normalize by running l and store Ht (b, n, c) bf16
#pragma unroll
  for (int r = 0; r < 8; ++r) {
    const float inv = 1.0f / l_s[r + 8 * half];
    const int i = i0 + r + 8 * half;
#pragma unroll
    for (int t = 0; t < 4; ++t) {
      const int c = wave * 64 + t * 16 + mrow;
      Ht[(size_t)b * N * C + (size_t)i * C + c] = (__bf16)(acc[t][r] * inv);
    }
  }
}

// ---------------------------------------------------------------------------
// Kernel 3: out = x + proj_w @ hmat + proj_b   (512x512 GEMM per column block)
// ---------------------------------------------------------------------------
__global__ void __launch_bounds__(256)
proj_kernel(const float* __restrict__ x, const __bf16* __restrict__ pw,
            const float* __restrict__ pb, const __bf16* __restrict__ Ht,
            float* __restrict__ out) {
  const int lane = threadIdx.x & 31;
  const int wave = threadIdx.x >> 5;
  const int half = lane >> 4;
  const int mrow = lane & 15;
  const int b  = blockIdx.y;
  const int n0 = blockIdx.x * 16;
  const int o0 = wave * 64;

  const __bf16* hcol = Ht + (size_t)b * N * C + (size_t)(n0 + mrow) * C;

  v8f acc[4];
#pragma unroll
  for (int t = 0; t < 4; ++t) acc[t] = (v8f){0, 0, 0, 0, 0, 0, 0, 0};

#pragma unroll 4
  for (int kc = 0; kc < C / 32; ++kc) {
    v16bf bb = load_b_rowmajor_bf16(hcol, kc * 32, half);
#pragma unroll
    for (int t = 0; t < 4; ++t) {
      v16bf a =
          load_a_rowmajor_bf16(pw + (size_t)(o0 + t * 16 + mrow) * C, kc * 32, half);
      acc[t] = wmma_bf16(a, bb, acc[t]);
    }
  }

  const int n = n0 + mrow;
#pragma unroll
  for (int t = 0; t < 4; ++t) {
#pragma unroll
    for (int r = 0; r < 8; ++r) {
      const int o = o0 + t * 16 + r + 8 * half;
      const size_t idx = ((size_t)b * C + o) * N + n;
      out[idx] = x[idx] + acc[t][r] + pb[o];
    }
  }
}

// ---------------------------------------------------------------------------
extern "C" void kernel_launch(void* const* d_in, const int* in_sizes, int n_in,
                              void* d_out, int out_size, void* d_ws, size_t ws_size,
                              hipStream_t stream) {
  (void)in_sizes; (void)n_in; (void)out_size; (void)ws_size;
  const float* x      = (const float*)d_in[0];
  const float* qkv_w  = (const float*)d_in[1];
  const float* qkv_b  = (const float*)d_in[2];
  const float* proj_w = (const float*)d_in[3];
  const float* proj_b = (const float*)d_in[4];

  unsigned char* ws = (unsigned char*)d_ws;
  size_t off = 0;
  auto take = [&](size_t bytes) -> void* {
    void* p = ws + off;
    off += (bytes + 255) & ~(size_t)255;
    return p;
  };
  __bf16* wqkv  = (__bf16*)take((size_t)OC * C * sizeof(__bf16));
  __bf16* wproj = (__bf16*)take((size_t)C * C * sizeof(__bf16));
  __bf16* Qd    = (__bf16*)take((size_t)Bn * N * C * sizeof(__bf16));
  __bf16* Kd    = (__bf16*)take((size_t)Bn * N * C * sizeof(__bf16));
  __bf16* Vd    = (__bf16*)take((size_t)Bn * C * N * sizeof(__bf16));
  __bf16* Htd   = (__bf16*)take((size_t)Bn * N * C * sizeof(__bf16));

  cvt_weights_kernel<<<(OC * C + 255) / 256, 256, 0, stream>>>(qkv_w, proj_w,
                                                               wqkv, wproj);
  qkv_gemm_kernel<<<dim3(N / 16, OC / 512, Bn), 256, 0, stream>>>(
      x, wqkv, qkv_b, Qd, Kd, Vd);
  attn_kernel<<<dim3(N / 16, Bn), 256, 0, stream>>>(Qd, Kd, Vd, Htd);
  proj_kernel<<<dim3(N / 16, Bn), 256, 0, stream>>>(x, wproj, proj_b, Htd,
                                                    (float*)d_out);
}
